// DXVAE_67800353735289
// MI455X (gfx1250) — matrix operand order repaired and checked by reading, not compiled, not run on hardware
//
#include <hip/hip_runtime.h>
#include <hip/hip_bf16.h>
#include <math.h>

// ---------------------------------------------------------------------------
// Types for WMMA fragments (CDNA5 gfx1250, wave32)
// ---------------------------------------------------------------------------
typedef __attribute__((ext_vector_type(16))) __bf16        v16bf;
typedef __attribute__((ext_vector_type(8)))  float         v8f;
typedef __attribute__((ext_vector_type(4)))  unsigned int  u32x4;

union Frag {
  v16bf          v;      // operand for v_wmma_f32_16x16x32_bf16
  unsigned short u[16];  // raw bf16 bits
  u32x4          q[2];   // two 16B vector loads
};

__device__ __forceinline__ unsigned short f2bf(float f) {
  unsigned int x = __float_as_uint(f);
  x += 0x7FFFu + ((x >> 16) & 1u);   // round-to-nearest-even
  return (unsigned short)(x >> 16);
}
__device__ __forceinline__ float sigmoidf_(float x) {
  return 1.f / (1.f + __expf(-x));
}

// ---------------------------------------------------------------------------
// Weight repack: src [N x K] f32 row-major -> dst [N x Kpad] bf16, zero padded.
// A W row is a B-matrix column of x @ W^T -> B-fragment loads are contiguous.
// ---------------------------------------------------------------------------
__global__ void k_cvt_w(unsigned short* __restrict__ dst,
                        const float* __restrict__ src,
                        int N, int K, int Kpad) {
  int idx = blockIdx.x * blockDim.x + threadIdx.x;
  if (idx >= N * Kpad) return;
  int k = idx % Kpad, n = idx / Kpad;
  dst[idx] = (k < K) ? f2bf(src[(size_t)n * K + k]) : (unsigned short)0;
}

// ---------------------------------------------------------------------------
// Pack X[:, v, :Kact] -> bf16 [B x 32] zero-padded; optional second copy with
// rows masked by adj[b,v,v] (self-loop GRU input) in the same pass.
// ---------------------------------------------------------------------------
__global__ void k_pack_x(unsigned short* __restrict__ dstP,
                         unsigned short* __restrict__ dstM,
                         const float* __restrict__ X,
                         const int* __restrict__ adj,
                         int v, int Kact, int total) {
  int idx = blockIdx.x * blockDim.x + threadIdx.x;
  if (idx >= total) return;
  int j = idx & 31;
  int b = idx >> 5;
  unsigned short us = (j < Kact) ? f2bf(X[(size_t)b * 189 + v * 27 + j])
                                 : (unsigned short)0;
  dstP[idx] = us;
  if (dstM) dstM[idx] = (adj[b * 49 + 8 * v] > 0) ? us : (unsigned short)0;
}

// ---------------------------------------------------------------------------
// Hcat build: Hcat[(b*k + kk), 0:2048] = [pred? h(node):0 , succ? h(node):0]
// emitted as bf16 bits to feed WMMA directly.
// ---------------------------------------------------------------------------
__global__ void k_build_hcat(unsigned short* __restrict__ hcat,
                             const float* __restrict__ hidden,
                             const int* __restrict__ adj,
                             int v, int kcnt, int total) {
  int idx = blockIdx.x * blockDim.x + threadIdx.x;
  if (idx >= total) return;
  const int H = 1024, H2 = 2048;
  int j   = idx % H2;
  int row = idx / H2;
  int kk  = row % kcnt;
  int b   = row / kcnt;
  int node = v + 1 + kk;
  int msk = (j < H) ? adj[b * 49 + node * 7 + v]    // pred mask
                    : adj[b * 49 + v * 7 + node];   // succ mask
  float hval = hidden[((size_t)b * 7 + node) * H + (j & (H - 1))];
  hcat[idx] = (msk > 0) ? f2bf(hval) : (unsigned short)0;
}

// H_in = sum_kk sigmoid(g_lin)*m_lin ; writes f32 (for GRU hprev) + bf16 (GEMM A)
__global__ void k_gate_reduce(float* __restrict__ Hin,
                              unsigned short* __restrict__ bHin,
                              const float* __restrict__ g_lin,
                              const float* __restrict__ m_lin,
                              int kcnt, int total) {
  int idx = blockIdx.x * blockDim.x + threadIdx.x;
  if (idx >= total) return;
  int j = idx & 1023;
  int b = idx >> 10;
  float s = 0.f;
  for (int kk = 0; kk < kcnt; ++kk) {
    size_t r = ((size_t)b * kcnt + kk) * 1024 + j;
    s += sigmoidf_(g_lin[r]) * m_lin[r];
  }
  Hin[idx]  = s;
  bHin[idx] = f2bf(s);
}

// GRU combine (biases fused in GEMM epilogue); writes strided f32 + flat bf16
__global__ void k_gru_combine(float* __restrict__ out, int outStride,
                              unsigned short* __restrict__ outBf,
                              const float* __restrict__ gi,
                              const float* __restrict__ gh,
                              const float* __restrict__ hprev,
                              int total) {
  int idx = blockIdx.x * blockDim.x + threadIdx.x;
  if (idx >= total) return;
  int j = idx & 1023;
  int b = idx >> 10;
  size_t base = (size_t)b * 3072;
  float r = sigmoidf_(gi[base + j]        + gh[base + j]);
  float z = sigmoidf_(gi[base + 1024 + j] + gh[base + 1024 + j]);
  float n = tanhf(gi[base + 2048 + j] + r * gh[base + 2048 + j]);
  float h = (1.f - z) * n + z * hprev[(size_t)b * 1024 + j];
  out[(size_t)b * outStride + j] = h;
  outBf[idx] = f2bf(h);
}

__global__ void k_softplus(float* __restrict__ x, int total) {
  int idx = blockIdx.x * blockDim.x + threadIdx.x;
  if (idx >= total) return;
  float v = x[idx];
  x[idx] = (v > 20.f) ? v : log1pf(__expf(v));
}

// ---------------------------------------------------------------------------
// WMMA GEMM:  C[M x N] = A[M x Kpad] @ W^T (+ bias[N])
//   A : bf16 [M x Kpad] row-major (pre-packed/padded by producer kernels)
//   W : bf16 [N x Kpad] row-major (= B columns contiguous)
// Block = 128 thr (4 waves); wave tile = 16(M) x 64(N); K step = 32.
// K loop unrolled x2 with two FIXED ping-pong tile buffers: no register
// rotation copies (round 2 showed a 40x v_dual_mov block per iteration),
// and each buffer's load clause overlaps the other buffer's 4-wmma group.
// Fragment layouts used here (wave32; A/B k order is contiguous, not the ISA table order):
//   A 16x32 : lane=M (l&15), 16 contiguous K starting at kb + 16*(lane>>4)
//
//   B 32x16 : lane=N (l&15), 16 contiguous K starting at kb + 16*(lane>>4)
//   C 16x16 : vgpr r -> M = r + 8*(lane>>4), N = lane&15
// ---------------------------------------------------------------------------
struct Tile { Frag a, b0, b1, b2, b3; };

__global__ __launch_bounds__(128)
void k_gemm_bf16(const unsigned short* __restrict__ A, int Kpad,
                 const unsigned short* __restrict__ Wb,
                 const float* __restrict__ bias,
                 float* __restrict__ C, int ldc) {
  const int lane = threadIdx.x & 31;
  const int wave = threadIdx.x >> 5;
  const int half = lane >> 4;
  const int l16  = lane & 15;
  const long mBase = (long)blockIdx.y * 64 + wave * 16;
  const long nBase = (long)blockIdx.x * 64;

  const unsigned short* Ar = A  + (mBase + l16) * (long)Kpad + 16 * half;
  const unsigned short* B0 = Wb + (nBase + l16) * (long)Kpad + 16 * half;
  const unsigned short* B1 = B0 + 16 * (long)Kpad;
  const unsigned short* B2 = B1 + 16 * (long)Kpad;
  const unsigned short* B3 = B2 + 16 * (long)Kpad;

  v8f acc0 = {0.f,0.f,0.f,0.f,0.f,0.f,0.f,0.f};
  v8f acc1 = acc0, acc2 = acc0, acc3 = acc0;

  auto ld = [&](Tile& t, int kb) {
    t.a.q[0]  = *(const u32x4*)(Ar + kb);
    t.a.q[1]  = *(const u32x4*)(Ar + kb + 8);
    t.b0.q[0] = *(const u32x4*)(B0 + kb);
    t.b0.q[1] = *(const u32x4*)(B0 + kb + 8);
    t.b1.q[0] = *(const u32x4*)(B1 + kb);
    t.b1.q[1] = *(const u32x4*)(B1 + kb + 8);
    t.b2.q[0] = *(const u32x4*)(B2 + kb);
    t.b2.q[1] = *(const u32x4*)(B2 + kb + 8);
    t.b3.q[0] = *(const u32x4*)(B3 + kb);
    t.b3.q[1] = *(const u32x4*)(B3 + kb + 8);
  };
  auto mm = [&](const Tile& t) {
    acc0 = __builtin_amdgcn_wmma_f32_16x16x32_bf16(false, t.a.v, false, t.b0.v,
                                                   (short)0, acc0, false, false);
    acc1 = __builtin_amdgcn_wmma_f32_16x16x32_bf16(false, t.a.v, false, t.b1.v,
                                                   (short)0, acc1, false, false);
    acc2 = __builtin_amdgcn_wmma_f32_16x16x32_bf16(false, t.a.v, false, t.b2.v,
                                                   (short)0, acc2, false, false);
    acc3 = __builtin_amdgcn_wmma_f32_16x16x32_bf16(false, t.a.v, false, t.b3.v,
                                                   (short)0, acc3, false, false);
  };

  Tile t0, t1;
  if (Kpad == 32) {                 // single-tile GEMMs (x @ Wih^T)
    ld(t0, 0);
    mm(t0);
  } else {                          // tile count is even (Kpad = 1024 or 2048)
    ld(t0, 0);
    ld(t1, 32);
    int kb = 0;
    for (; kb + 64 < Kpad; kb += 64) {
      mm(t0); ld(t0, kb + 64);      // wmma group overlaps the other buffer's loads
      mm(t1); ld(t1, kb + 96);
    }
    mm(t0);
    mm(t1);
  }

  const long rowB = mBase + 8 * half;
  auto st = [&](const v8f& acc, int t) {
    const long col = nBase + t * 16 + l16;
    const float bv = bias ? bias[col] : 0.f;
#pragma unroll
    for (int r = 0; r < 8; ++r)
      C[(rowB + r) * (long)ldc + col] = acc[r] + bv;
  };
  st(acc0, 0); st(acc1, 1); st(acc2, 2); st(acc3, 3);
}

// ---------------------------------------------------------------------------
// Host orchestration
// ---------------------------------------------------------------------------
extern "C" void kernel_launch(void* const* d_in, const int* in_sizes, int n_in,
                              void* d_out, int out_size, void* d_ws, size_t ws_size,
                              hipStream_t stream) {
  (void)in_sizes; (void)n_in; (void)out_size; (void)ws_size;
  const int H = 1024, B = 2048, NN = 7, Xd = 27, X0d = 23, Zd = 256;
  const int H2 = 2 * H, H3 = 3 * H;

  const float* X      = (const float*)d_in[0];
  const int*   adj    = (const int*)  d_in[1];
  const float* Wih_c  = (const float*)d_in[2];
  const float* Whh_c  = (const float*)d_in[3];
  const float* bih_c  = (const float*)d_in[4];
  const float* bhh_c  = (const float*)d_in[5];
  const float* Wih_l  = (const float*)d_in[6];
  const float* Whh_l  = (const float*)d_in[7];
  const float* bih_l  = (const float*)d_in[8];
  const float* bhh_l  = (const float*)d_in[9];
  const float* Wih_r  = (const float*)d_in[10];
  const float* Whh_r  = (const float*)d_in[11];
  const float* bih_r  = (const float*)d_in[12];
  const float* bhh_r  = (const float*)d_in[13];
  const float* Wg     = (const float*)d_in[14];
  const float* bg     = (const float*)d_in[15];
  const float* Wm     = (const float*)d_in[16];
  const float* Wmu    = (const float*)d_in[17];
  const float* bmu    = (const float*)d_in[18];
  const float* Wstd   = (const float*)d_in[19];
  const float* bstd   = (const float*)d_in[20];

  // ---- workspace carving (256B aligned slices) ----
  char* p = (char*)d_ws;
  auto take = [&](size_t bytes) -> char* {
    char* r = p;
    p += (bytes + 255) & ~(size_t)255;
    return r;
  };
  float*          hidden = (float*)         take((size_t)B * NN * H * 4);
  unsigned short* hcat   = (unsigned short*)take((size_t)B * 6 * H2 * 2);
  float*          C1     = (float*)         take((size_t)B * 6 * H * 4); // >= B*3H*4
  float*          C2     = (float*)         take((size_t)B * 6 * H * 4);
  float*          Hin    = (float*)         take((size_t)B * H * 4);
  float*          Htmp   = (float*)         take((size_t)B * H * 4);
  unsigned short* bHin   = (unsigned short*)take((size_t)B * H * 2);
  unsigned short* bHtmp  = (unsigned short*)take((size_t)B * H * 2);
  unsigned short* bHg    = (unsigned short*)take((size_t)B * H * 2);
  unsigned short* Xp     = (unsigned short*)take((size_t)B * 32 * 2);
  unsigned short* Xl     = (unsigned short*)take((size_t)B * 32 * 2);
  unsigned short* bWih_c = (unsigned short*)take((size_t)H3 * 32 * 2);
  unsigned short* bWhh_c = (unsigned short*)take((size_t)H3 * H * 2);
  unsigned short* bWih_l = (unsigned short*)take((size_t)H3 * 32 * 2);
  unsigned short* bWhh_l = (unsigned short*)take((size_t)H3 * H * 2);
  unsigned short* bWih_r = (unsigned short*)take((size_t)H3 * 32 * 2);
  unsigned short* bWhh_r = (unsigned short*)take((size_t)H3 * H * 2);
  unsigned short* bWg    = (unsigned short*)take((size_t)H  * H2 * 2);
  unsigned short* bWm    = (unsigned short*)take((size_t)H  * H2 * 2);
  unsigned short* bWmu   = (unsigned short*)take((size_t)Zd * H * 2);
  unsigned short* bWstd  = (unsigned short*)take((size_t)Zd * H * 2);

  const int EB = 256;
  auto cvt = [&](unsigned short* dst, const float* src, int N, int K, int Kpad) {
    int total = N * Kpad;
    k_cvt_w<<<(total + EB - 1) / EB, EB, 0, stream>>>(dst, src, N, K, Kpad);
  };
  cvt(bWih_c, Wih_c, H3, Xd,  32);
  cvt(bWhh_c, Whh_c, H3, H,   H);
  cvt(bWih_l, Wih_l, H3, Xd,  32);
  cvt(bWhh_l, Whh_l, H3, H,   H);
  cvt(bWih_r, Wih_r, H3, X0d, 32);
  cvt(bWhh_r, Whh_r, H3, H,   H);
  cvt(bWg,    Wg,    H,  H2,  H2);
  cvt(bWm,    Wm,    H,  H2,  H2);
  cvt(bWmu,   Wmu,   Zd, H,   H);
  cvt(bWstd,  Wstd,  Zd, H,   H);

  auto gemm = [&](const unsigned short* A, int M, int Kpad,
                  const unsigned short* Wb, int N, const float* bias,
                  float* Cp, int ldc) {
    dim3 g(N / 64, M / 64);
    k_gemm_bf16<<<g, 128, 0, stream>>>(A, Kpad, Wb, bias, Cp, ldc);
  };

  // h = 0 for node v = NN-1, processed first with k = 0 (f32 + bf16 views)
  hipMemsetAsync(Hin,  0, (size_t)B * H * sizeof(float), stream);
  hipMemsetAsync(bHin, 0, (size_t)B * H * sizeof(unsigned short), stream);

  for (int v = NN - 1; v >= 0; --v) {
    const int k = (NN - 1) - v;
    if (k > 0) {
      // ---- neighbor gating: H_in = sum_k sigmoid(Hcat Wg^T + bg) * (Hcat Wm^T)
      int tot = B * k * H2;
      k_build_hcat<<<(tot + EB - 1) / EB, EB, 0, stream>>>(hcat, hidden, adj, v, k, tot);
      gemm(hcat, B * k, H2, bWg, H, bg, C1, H);       // g_lin
      gemm(hcat, B * k, H2, bWm, H, nullptr, C2, H);  // m_lin
      int th = B * H;
      k_gate_reduce<<<(th + EB - 1) / EB, EB, 0, stream>>>(Hin, bHin, C1, C2, k, th);
    }
    const int th = B * H;
    const int tx = B * 32;
    if (v == 0) {
      // ---- root GRU (X0d input features)
      k_pack_x<<<(tx + EB - 1) / EB, EB, 0, stream>>>(Xp, nullptr, X, adj, v, X0d, tx);
      gemm(Xp,   B, 32, bWih_r, H3, bih_r, C1, H3);
      gemm(bHin, B, H,  bWhh_r, H3, bhh_r, C2, H3);
      k_gru_combine<<<(th + EB - 1) / EB, EB, 0, stream>>>(hidden, NN * H, bHg,
                                                           C1, C2, Hin, th);
    } else {
      // ---- cell GRU + self-loop GRU (x rows pre-masked by adj[b,v,v])
      k_pack_x<<<(tx + EB - 1) / EB, EB, 0, stream>>>(Xp, Xl, X, adj, v, Xd, tx);
      gemm(Xp,   B, 32, bWih_c, H3, bih_c, C1, H3);
      gemm(bHin, B, H,  bWhh_c, H3, bhh_c, C2, H3);
      k_gru_combine<<<(th + EB - 1) / EB, EB, 0, stream>>>(Htmp, H, bHtmp,
                                                           C1, C2, Hin, th);
      gemm(Xl,    B, 32, bWih_l, H3, bih_l, C1, H3);
      gemm(bHtmp, B, H,  bWhh_l, H3, bhh_l, C2, H3);
      k_gru_combine<<<(th + EB - 1) / EB, EB, 0, stream>>>(hidden + (size_t)v * H,
                                                           NN * H, bHg, C1, C2, Htmp, th);
    }
  }

  // ---- head: mu and softplus(std) from hidden[:,0,:] (bf16 copy in bHg)
  float* mu   = (float*)d_out;
  float* std_ = (float*)d_out + (size_t)B * Zd;
  gemm(bHg, B, H, bWmu,  Zd, bmu,  mu,   Zd);
  gemm(bHg, B, H, bWstd, Zd, bstd, std_, Zd);
  int th = B * Zd;
  k_softplus<<<(th + EB - 1) / EB, EB, 0, stream>>>(std_, th);
}
